// Mamba2_22204980920821
// MI455X (gfx1250) — matrix operand order repaired and checked by reading, not compiled, not run on hardware
//
#include <hip/hip_runtime.h>
#include <math.h>

typedef __attribute__((ext_vector_type(16))) _Float16 v16h;
typedef __attribute__((ext_vector_type(8)))  _Float16 v8h;
typedef __attribute__((ext_vector_type(4)))  _Float16 v4h;
typedef __attribute__((ext_vector_type(8)))  float    v8f;

#define BATCH  32
#define SEQ    2048
#define IN_DIM 32
#define HID    512
#define STATE  64
#define NROWS  (BATCH * SEQ)   // 65536

// ---------- WMMA helpers ----------

__device__ __forceinline__ v8f wmma_f16(v16h a, v16h b, v8f c) {
    // D = A(16x32 f16) * B(32x16 f16) + C(16x16 f32)
    return __builtin_amdgcn_wmma_f32_16x16x32_f16(
        /*neg_a=*/false, a, /*neg_b=*/false, b,
        /*c_mod=*/(short)0, c, /*reuse_a=*/false, /*reuse_b=*/false);
}

// Load 16x32 f16 A-operand tile from row-major f16 matrix.
// CDNA5 layout: lane m=L%16 holds row m; lanes<16: K {0..7,16..23}, lanes>=16: K {8..15,24..31}
__device__ __forceinline__ v16h load_a16(const _Float16* __restrict__ base, int lda,
                                         int rbase, int k0, int lane) {
    int m  = lane & 15;
    int kb = (lane >> 4) << 3;              // 0 or 8
    const _Float16* p = base + (size_t)(rbase + m) * lda + k0 + kb;
    v8h lo = *(const v8h*)p;                // K = k0+kb .. +7      (16B load)
    v8h hi = *(const v8h*)(p + 16);         // K = k0+16+kb .. +7   (16B load)
    return __builtin_shufflevector(lo, hi, 0,1,2,3,4,5,6,7,8,9,10,11,12,13,14,15);
}

// ---------- elementwise / pack kernels ----------

__global__ __launch_bounds__(256) void cvt_f16_kernel(const float* __restrict__ src,
                                                      _Float16* __restrict__ dst, int n) {
    int i = blockIdx.x * 256 + threadIdx.x;
    if (i < n) dst[i] = (_Float16)src[i];
}

// Pack src (K x N f32, element (k,n) at src[k*rs + n*cs]) into WMMA B-operand layout:
// tile = kt*NT + nt; per lane 16 contiguous f16: j -> K = kt*32 + ((j&8)<<1) + ((lane&16)>>1) + (j&7), N = nt*16 + (lane&15)
__global__ __launch_bounds__(256) void pack_b_kernel(const float* __restrict__ src,
                                                     _Float16* __restrict__ dst,
                                                     int KT, int NT, int rs, int cs) {
    int t    = blockIdx.x * 256 + threadIdx.x;
    int lane = t & 31;
    int tile = t >> 5;
    if (tile >= KT * NT) return;
    int kt = tile / NT, nt = tile % NT;
    int kb = lane & 16;                     // 0 or 16
    int n  = nt * 16 + (lane & 15);
    _Float16* o = dst + ((size_t)tile * 32 + lane) * 16;
#pragma unroll
    for (int j = 0; j < 16; ++j) {
        int k = kt * 32 + ((j & 8) << 1) + (kb >> 1) + (j & 7);
        o[j] = (_Float16)src[(size_t)k * rs + (size_t)n * cs];
    }
}

// LayerNorm over HID=512; one wave per row; output f16.
__global__ __launch_bounds__(256) void ln_kernel(const float* __restrict__ src,
                                                 const float* __restrict__ g,
                                                 const float* __restrict__ b,
                                                 _Float16* __restrict__ dst) {
    int wave = threadIdx.x >> 5;
    int lane = threadIdx.x & 31;
    int row  = blockIdx.x * 8 + wave;
    const float4* s4 = (const float4*)(src + (size_t)row * HID);
    float4 v[4];
    float s = 0.f, ss = 0.f;
#pragma unroll
    for (int i = 0; i < 4; ++i) {
        v[i] = s4[i * 32 + lane];
        s  += v[i].x + v[i].y + v[i].z + v[i].w;
        ss += v[i].x * v[i].x + v[i].y * v[i].y + v[i].z * v[i].z + v[i].w * v[i].w;
    }
#pragma unroll
    for (int off = 16; off > 0; off >>= 1) {
        s  += __shfl_xor(s,  off, 32);
        ss += __shfl_xor(ss, off, 32);
    }
    float mean = s * (1.0f / HID);
    float var  = ss * (1.0f / HID) - mean * mean;
    float rstd = rsqrtf(var + 1e-5f);
#pragma unroll
    for (int i = 0; i < 4; ++i) {
        int c = (i * 32 + lane) * 4;
        v4h o;
        o[0] = (_Float16)((v[i].x - mean) * rstd * g[c + 0] + b[c + 0]);
        o[1] = (_Float16)((v[i].y - mean) * rstd * g[c + 1] + b[c + 1]);
        o[2] = (_Float16)((v[i].z - mean) * rstd * g[c + 2] + b[c + 2]);
        o[3] = (_Float16)((v[i].w - mean) * rstd * g[c + 3] + b[c + 3]);
        *(v4h*)(dst + (size_t)row * HID + c) = o;
    }
}

// ---------- GEMM kernels (WMMA f16 -> f32) ----------

// h = x(f16, 65536x32) @ Win + bin ; KT=1, wave does 16 rows x 64 cols
__global__ __launch_bounds__(256) void proj_kernel(const _Float16* __restrict__ xh,
                                                   const _Float16* __restrict__ wpack,
                                                   const float* __restrict__ bias,
                                                   float* __restrict__ out) {
    int wave = threadIdx.x >> 5, lane = threadIdx.x & 31;
    int rbase = blockIdx.x * 16;
    v16h a = load_a16(xh, IN_DIM, rbase, 0, lane);
    v8f acc[4];
#pragma unroll
    for (int nt = 0; nt < 4; ++nt) {
        int tile = wave * 4 + nt;   // NT = 32
        v16h bb = *(const v16h*)(wpack + ((size_t)tile * 32 + lane) * 16);
        v8f z = {};
        acc[nt] = wmma_f16(a, bb, z);
    }
    int n = lane & 15, mo = (lane >> 4) << 3;
#pragma unroll
    for (int nt = 0; nt < 4; ++nt) {
        int c = wave * 64 + nt * 16 + n;
        float bv = bias[c];
#pragma unroll
        for (int vv = 0; vv < 8; ++vv)
            out[(size_t)(rbase + vv + mo) * HID + c] = acc[nt][vv] + bv;
    }
}

// UA = xn(f16, 65536x512) @ A(512x64) ; KT=16, NT=4; wave does 16 rows x all 64 cols
__global__ __launch_bounds__(256) void ua_kernel(const _Float16* __restrict__ xn,
                                                 const _Float16* __restrict__ apack,
                                                 float* __restrict__ ua) {
    int wave = threadIdx.x >> 5, lane = threadIdx.x & 31;
    int rbase = blockIdx.x * 128 + wave * 16;
    v8f acc[4] = {};
    for (int kt = 0; kt < 16; ++kt) {
        v16h a = load_a16(xn, HID, rbase, kt * 32, lane);
#pragma unroll
        for (int nt = 0; nt < 4; ++nt) {
            v16h bb = *(const v16h*)(apack + ((size_t)(kt * 4 + nt) * 32 + lane) * 16);
            acc[nt] = wmma_f16(a, bb, acc[nt]);
        }
    }
    int n = lane & 15, mo = (lane >> 4) << 3;
#pragma unroll
    for (int nt = 0; nt < 4; ++nt)
#pragma unroll
        for (int vv = 0; vv < 8; ++vv)
            ua[(size_t)(rbase + vv + mo) * STATE + nt * 16 + n] = acc[nt][vv];
}

// 2048 independent scalar recurrences h = tanh(ua + h); one thread per (batch, state)
__global__ __launch_bounds__(256) void scan_kernel(const float* __restrict__ ua,
                                                   _Float16* __restrict__ Hs) {
    int gid = blockIdx.x * 256 + threadIdx.x;
    if (gid >= BATCH * STATE) return;
    int b = gid >> 6, s = gid & 63;
    size_t base = (size_t)b * SEQ * STATE + s;
    float h = 0.f;
    for (int t = 0; t < SEQ; ++t) {
        size_t idx = base + (size_t)t * STATE;
        h = tanhf(ua[idx] + h);
        Hs[idx] = (_Float16)h;
    }
}

// h = H(f16, 65536x64) @ C^T(64x512) + xn*D + h  (fused residual); KT=2, NT=32
__global__ __launch_bounds__(256) void yc_kernel(const _Float16* __restrict__ Hs,
                                                 const _Float16* __restrict__ cpack,
                                                 const _Float16* __restrict__ xn,
                                                 const float* __restrict__ Dv,
                                                 float* __restrict__ h) {
    int wave = threadIdx.x >> 5, lane = threadIdx.x & 31;
    int rbase = blockIdx.x * 16;
    v8f acc[4] = {};
#pragma unroll
    for (int kt = 0; kt < 2; ++kt) {
        v16h a = load_a16(Hs, STATE, rbase, kt * 32, lane);
#pragma unroll
        for (int ntl = 0; ntl < 4; ++ntl) {
            int tile = kt * 32 + wave * 4 + ntl;
            v16h bb = *(const v16h*)(cpack + ((size_t)tile * 32 + lane) * 16);
            acc[ntl] = wmma_f16(a, bb, acc[ntl]);
        }
    }
    int n = lane & 15, mo = (lane >> 4) << 3;
#pragma unroll
    for (int ntl = 0; ntl < 4; ++ntl) {
        int c = wave * 64 + ntl * 16 + n;
        float d = Dv[c];
#pragma unroll
        for (int vv = 0; vv < 8; ++vv) {
            size_t off = (size_t)(rbase + vv + mo) * HID + c;
            h[off] = acc[ntl][vv] + (float)xn[off] * d + h[off];
        }
    }
}

// Final LN (last timestep only) + MLP head with exact GELU. One block per batch row.
__global__ __launch_bounds__(256) void head_kernel(const float* __restrict__ h,
                                                   const float* __restrict__ ng,
                                                   const float* __restrict__ nb,
                                                   const float* __restrict__ W1,
                                                   const float* __restrict__ b1,
                                                   const float* __restrict__ W2,
                                                   const float* __restrict__ b2,
                                                   float* __restrict__ out) {
    __shared__ float xs[HID];
    __shared__ float y1[256];
    __shared__ float red[20];
    int b = blockIdx.x;
    const float* row = h + ((size_t)b * SEQ + (SEQ - 1)) * HID;
    int tid = threadIdx.x;
    float a0 = row[tid], a1 = row[tid + 256];
    float s = a0 + a1, ss = a0 * a0 + a1 * a1;
#pragma unroll
    for (int off = 16; off > 0; off >>= 1) {
        s  += __shfl_xor(s,  off, 32);
        ss += __shfl_xor(ss, off, 32);
    }
    if ((tid & 31) == 0) { red[tid >> 5] = s; red[8 + (tid >> 5)] = ss; }
    __syncthreads();
    if (tid == 0) {
        float ts = 0.f, tss = 0.f;
        for (int i = 0; i < 8; ++i) { ts += red[i]; tss += red[8 + i]; }
        float mean = ts * (1.0f / HID);
        float var  = tss * (1.0f / HID) - mean * mean;
        red[16] = mean;
        red[17] = rsqrtf(var + 1e-5f);
    }
    __syncthreads();
    float mean = red[16], rstd = red[17];
    xs[tid]       = (a0 - mean) * rstd * ng[tid]       + nb[tid];
    xs[tid + 256] = (a1 - mean) * rstd * ng[tid + 256] + nb[tid + 256];
    __syncthreads();
    float acc = b1[tid];
    for (int k = 0; k < HID; ++k) acc += xs[k] * W1[(size_t)k * 256 + tid];
    y1[tid] = 0.5f * acc * (1.0f + erff(acc * 0.7071067811865475f));
    __syncthreads();
    if (tid < 2) {
        float o = b2[tid];
        for (int k = 0; k < 256; ++k) o += y1[k] * W2[k * 2 + tid];
        out[b * 2 + tid] = o;
    }
}

// ---------- launch ----------

extern "C" void kernel_launch(void* const* d_in, const int* in_sizes, int n_in,
                              void* d_out, int out_size, void* d_ws, size_t ws_size,
                              hipStream_t stream) {
    (void)in_sizes; (void)n_in; (void)out_size; (void)ws_size;
    const float* x      = (const float*)d_in[0];
    const float* Win    = (const float*)d_in[1];
    const float* bin_   = (const float*)d_in[2];
    const float* A      = (const float*)d_in[3];
    const float* C      = (const float*)d_in[4];
    const float* Dvec   = (const float*)d_in[5];
    const float* ln_g   = (const float*)d_in[6];
    const float* ln_b   = (const float*)d_in[7];
    const float* norm_g = (const float*)d_in[8];
    const float* norm_b = (const float*)d_in[9];
    const float* W1     = (const float*)d_in[10];
    const float* b1     = (const float*)d_in[11];
    const float* W2     = (const float*)d_in[12];
    const float* b2     = (const float*)d_in[13];

    char* ws = (char*)d_ws;
    size_t off = 0;
    auto alloc = [&](size_t bytes) -> void* {
        void* p = ws + off;
        off = (off + bytes + 255) & ~(size_t)255;
        return p;
    };
    float*    h     = (float*)   alloc((size_t)NROWS * HID * 4);     // 128 MB
    _Float16* xn    = (_Float16*)alloc((size_t)NROWS * HID * 2);     //  64 MB
    float*    ua    = (float*)   alloc((size_t)NROWS * STATE * 4);   //  16 MB
    _Float16* Hs    = (_Float16*)alloc((size_t)NROWS * STATE * 2);   //   8 MB
    _Float16* xh    = (_Float16*)alloc((size_t)NROWS * IN_DIM * 2);  //   4 MB
    _Float16* wpack = (_Float16*)alloc((size_t)IN_DIM * HID * 2);
    _Float16* apack = (_Float16*)alloc((size_t)HID * STATE * 2);
    _Float16* cpack = (_Float16*)alloc((size_t)HID * STATE * 2);

    // input conversion + weight pack + input projection
    cvt_f16_kernel<<<(NROWS * IN_DIM + 255) / 256, 256, 0, stream>>>(x, xh, NROWS * IN_DIM);
    pack_b_kernel<<<(1 * 32 * 32 + 255) / 256, 256, 0, stream>>>(Win, wpack, 1, 32, HID, 1);
    proj_kernel<<<NROWS / 16, 256, 0, stream>>>(xh, wpack, bin_, h);

    for (int l = 0; l < 4; ++l) {
        // pack A[l] (512x64, rs=STATE) and C[l]^T (64x512: elem(k,n)=C[n*STATE+k] -> rs=1, cs=STATE)
        pack_b_kernel<<<(16 * 4 * 32 + 255) / 256, 256, 0, stream>>>(
            A + (size_t)l * HID * STATE, apack, 16, 4, STATE, 1);
        pack_b_kernel<<<(2 * 32 * 32 + 255) / 256, 256, 0, stream>>>(
            C + (size_t)l * HID * STATE, cpack, 2, 32, 1, STATE);

        ln_kernel<<<NROWS / 8, 256, 0, stream>>>(h, ln_g + (size_t)l * HID, ln_b + (size_t)l * HID, xn);
        ua_kernel<<<NROWS / 128, 256, 0, stream>>>(xn, apack, ua);
        scan_kernel<<<(BATCH * STATE + 255) / 256, 256, 0, stream>>>(ua, Hs);
        yc_kernel<<<NROWS / 16, 256, 0, stream>>>(Hs, cpack, xn, Dvec + (size_t)l * HID, h);
    }

    head_kernel<<<BATCH, 256, 0, stream>>>(h, norm_g, norm_b, W1, b1, W2, b2, (float*)d_out);
}
